// TextAttention_74225624809554
// MI455X (gfx1250) — matrix operand order repaired and checked by reading, not compiled, not run on hardware
//
#include <hip/hip_runtime.h>
#include <hip/hip_bf16.h>

#define L_      4096
#define HS_     2560
#define H_      16
#define HKV_    8
#define DH_     256
#define WINDOW_ 1024
#define SOFTCAP_ 50.0f

typedef __attribute__((ext_vector_type(16))) _Float16 v16h;
typedef __attribute__((ext_vector_type(8)))  _Float16 v8h;
typedef __attribute__((ext_vector_type(4)))  _Float16 v4h;
typedef __attribute__((ext_vector_type(8)))  float    v8f;

static __device__ __forceinline__ v8f wmma_f16(v16h a, v16h b, v8f c) {
  // (neg_a, A, neg_b, B, c_mod, C, reuse_a, reuse_b)
  return __builtin_amdgcn_wmma_f32_16x16x32_f16(false, a, false, b, (short)0, c,
                                                false, false);
}

// A-fragment (16x32 f16, MxK): lane m = lane&15, g = lane>>4.
// halfs 0..15 -> k = 16g + 0..15
// Both runs are 16-byte aligned contiguous loads.
static __device__ __forceinline__ v16h load_a_frag(const _Float16* p, int g) {
  const v8h lo = *(const v8h*)(p + 16 * g);
  const v8h hi = *(const v8h*)(p + 16 * g + 8);
  return __builtin_shufflevector(lo, hi, 0, 1, 2, 3, 4, 5, 6, 7, 8, 9, 10, 11,
                                 12, 13, 14, 15);
}

// ---------------------------------------------------------------------------
// Bulk f32 -> f16 conversion (X and the four weight matrices), one pass.
// n must be a multiple of 4 (true for all our sizes).
// ---------------------------------------------------------------------------
__global__ __launch_bounds__(256) void cvt_f32_to_f16(
    const float* __restrict__ src, _Float16* __restrict__ dst, int n) {
  const int i = (blockIdx.x * 256 + threadIdx.x) * 4;
  if (i + 3 < n) {
    const float4 v = *(const float4*)(src + i);
    v4h o4;
    o4[0] = (_Float16)v.x;
    o4[1] = (_Float16)v.y;
    o4[2] = (_Float16)v.z;
    o4[3] = (_Float16)v.w;
    *(v4h*)(dst + i) = o4;
  }
}

// ---------------------------------------------------------------------------
// GEMM: Y[M,N] = A[M,K] * W[N,K]^T, all-f16 inputs, f32 accumulate.
// One wave computes a 64(M) x 64(N) tile: 4 A-frags x 4 B-frags -> 16 WMMAs
// per 32-wide K step, against 8 x 32-byte loads (b128 pairs).
// B-fragment per lane: n = lane&15; half h -> k = h + 16g (one v16h load).
// C-layout per lane: n = lane&15; float f -> m = f + 8g.
// ---------------------------------------------------------------------------
template <typename OutT>
__global__ __launch_bounds__(32) void gemm_f16(
    const _Float16* __restrict__ A, const _Float16* __restrict__ W,
    OutT* __restrict__ Y, int M, int N, int K) {
  const int lane = threadIdx.x & 31;
  const int g = lane >> 4, ln = lane & 15;
  const int m0 = blockIdx.y << 6;
  const int n0 = blockIdx.x << 6;
  const v8f vzero = {};
  v8f acc[4][4];
#pragma unroll
  for (int i = 0; i < 4; ++i)
#pragma unroll
    for (int j = 0; j < 4; ++j) acc[i][j] = vzero;

  const _Float16* arow[4];
  const _Float16* brow[4];
#pragma unroll
  for (int i = 0; i < 4; ++i)
    arow[i] = A + (size_t)(m0 + i * 16 + ln) * K;
#pragma unroll
  for (int j = 0; j < 4; ++j)
    brow[j] = W + (size_t)(n0 + j * 16 + ln) * K + 16 * g;

  for (int k0 = 0; k0 < K; k0 += 32) {
    __builtin_prefetch(arow[0] + k0 + 128, 0, 1);
    __builtin_prefetch(brow[0] + k0 + 128, 0, 1);
    // Issue all 16 fragment loads up front so the WMMA chain drains the
    // load counter progressively instead of waiting to zero each time.
    v16h a[4];
    v16h b[4];
#pragma unroll
    for (int i = 0; i < 4; ++i) a[i] = load_a_frag(arow[i] + k0, g);
#pragma unroll
    for (int j = 0; j < 4; ++j) b[j] = *(const v16h*)(brow[j] + k0);
#pragma unroll
    for (int j = 0; j < 4; ++j)
#pragma unroll
      for (int i = 0; i < 4; ++i) acc[i][j] = wmma_f16(a[i], b[j], acc[i][j]);
  }
#pragma unroll
  for (int i = 0; i < 4; ++i)
#pragma unroll
    for (int j = 0; j < 4; ++j)
#pragma unroll
      for (int f = 0; f < 8; ++f)
        Y[(size_t)(m0 + i * 16 + f + 8 * g) * N + n0 + j * 16 + ln] =
            (OutT)acc[i][j][f];
}

// ---------------------------------------------------------------------------
// Per-(token, head) RMS norm (+ weight) and RoPE; reads f16 raw q/k/v, writes:
//   Qf [H][L][DH] row-major, Kf [HKV][L][DH] row-major,
//   Vt [HKV][DH][L] transposed (PV B-fragments contiguous in key dim).
// blockIdx.x: 0..15 q-head, 16..23 k-head, 24..31 v-head; blockIdx.y = token.
// ---------------------------------------------------------------------------
__global__ __launch_bounds__(256) void normrope_kernel(
    const _Float16* __restrict__ qraw, const _Float16* __restrict__ kraw,
    const _Float16* __restrict__ vraw, const float* __restrict__ cosb,
    const float* __restrict__ sinb, const float* __restrict__ qw,
    const float* __restrict__ kw, _Float16* __restrict__ Qf,
    _Float16* __restrict__ Kf, _Float16* __restrict__ Vt) {
  __shared__ float sred[256];
  __shared__ float sval[256];
  const int t = threadIdx.x;
  const int idx = blockIdx.x;
  const int l = blockIdx.y;
  float x;
  if (idx < H_)
    x = (float)qraw[(size_t)l * (H_ * DH_) + idx * DH_ + t];
  else if (idx < H_ + HKV_)
    x = (float)kraw[(size_t)l * (HKV_ * DH_) + (idx - H_) * DH_ + t];
  else
    x = (float)vraw[(size_t)l * (HKV_ * DH_) + (idx - H_ - HKV_) * DH_ + t];
  sred[t] = x * x;
  __syncthreads();
#pragma unroll
  for (int s = 128; s > 0; s >>= 1) {
    if (t < s) sred[t] += sred[t + s];
    __syncthreads();
  }
  const float r = rsqrtf(sred[0] * (1.0f / DH_) + 1e-6f);
  float n = x * r;
  if (idx < H_) n *= qw[t];
  else if (idx < H_ + HKV_) n *= kw[t];
  sval[t] = n;
  __syncthreads();
  if (idx < H_ + HKV_) {
    const float c = cosb[(size_t)l * DH_ + t];
    const float s = sinb[(size_t)l * DH_ + t];
    const float rot = (t < DH_ / 2) ? -sval[t + DH_ / 2] : sval[t - DH_ / 2];
    const float o = n * c + rot * s;
    if (idx < H_)
      Qf[((size_t)idx * L_ + l) * DH_ + t] = (_Float16)o;
    else
      Kf[((size_t)(idx - H_) * L_ + l) * DH_ + t] = (_Float16)o;
  } else {
    const int kvh = idx - H_ - HKV_;
    Vt[((size_t)kvh * DH_ + t) * L_ + l] = (_Float16)n;
  }
}

// ---------------------------------------------------------------------------
// Flash-style sliding-window attention. One wave per (head, 16-query tile).
// S = Q K^T via WMMA; tanh softcap; analytic causal/window mask; online
// softmax (row stats via 16-lane shuffles); P transposed through LDS into the
// WMMA A layout; PV accumulated over all 16 DH tiles in registers.
// V fragments for the whole 32-key block are issued right after S is
// computed, so their latency is hidden under the softmax VALU section.
// ---------------------------------------------------------------------------
__global__ __launch_bounds__(32) void attn_kernel(
    const _Float16* __restrict__ Qf, const _Float16* __restrict__ Kf,
    const _Float16* __restrict__ Vt, _Float16* __restrict__ Oa) {
  const int lane = threadIdx.x & 31;
  const int g = lane >> 4, ln = lane & 15;
  const int h = blockIdx.y;
  const int kvh = h >> 1;  // H/HKV = 2
  const int q0 = blockIdx.x << 4;

  const _Float16* Qh = Qf + (size_t)h * L_ * DH_;
  const _Float16* Kh = Kf + (size_t)kvh * L_ * DH_;
  const _Float16* Vh = Vt + (size_t)kvh * DH_ * L_;

  // Preload the 16x256 Q tile as 8 A-fragments.
  v16h qa[8];
  {
    const _Float16* qrow = Qh + (size_t)(q0 + ln) * DH_;
#pragma unroll
    for (int tt = 0; tt < 8; ++tt) qa[tt] = load_a_frag(qrow + tt * 32, g);
  }

  const v8f vzero = {};
  v8f acc[16];
#pragma unroll
  for (int d = 0; d < 16; ++d) acc[d] = vzero;
  float mrow[8], lrow[8];
#pragma unroll
  for (int f = 0; f < 8; ++f) { mrow[f] = -1e30f; lrow[f] = 0.0f; }

  __shared__ alignas(64) _Float16 Pl[16][32];

  int kstart = q0 - (WINDOW_ - 1);
  if (kstart < 0) kstart = 0;
  kstart &= ~31;

  for (int kb = kstart; kb <= q0 + 15; kb += 32) {
    // ---- S = Q K^T for 32 keys (two 16-key sub-tiles) ----
    v8f S[2];
#pragma unroll
    for (int sub = 0; sub < 2; ++sub) {
      const _Float16* krow = Kh + (size_t)(kb + sub * 16 + ln) * DH_ + 16 * g;
      v16h kbf[8];
#pragma unroll
      for (int tt = 0; tt < 8; ++tt) kbf[tt] = *(const v16h*)(krow + tt * 32);
      v8f s = vzero;
#pragma unroll
      for (int tt = 0; tt < 8; ++tt) s = wmma_f16(qa[tt], kbf[tt], s);
      S[sub] = s;
    }

    // ---- Issue all V fragments for this key block now; they complete
    // ---- under the softmax VALU work below.
    v16h vb[16];
#pragma unroll
    for (int d = 0; d < 16; ++d) {
      const _Float16* vr = Vh + (size_t)(d * 16 + ln) * L_ + kb + 16 * g;
      vb[d] = *(const v16h*)(vr);
    }

    // ---- Softcap + causal sliding-window mask (C-layout) ----
#pragma unroll
    for (int sub = 0; sub < 2; ++sub)
#pragma unroll
      for (int f = 0; f < 8; ++f) {
        float s = S[sub][f];
        s = SOFTCAP_ * tanhf(s * (1.0f / SOFTCAP_));
        const int qrow = q0 + f + 8 * g;
        const int key = kb + sub * 16 + ln;
        const bool ok = (key <= qrow) && (qrow - key < WINDOW_);
        S[sub][f] = ok ? s : -1e30f;
      }

    // ---- Online softmax update over this 32-key block ----
    float scale_f[8];
#pragma unroll
    for (int f = 0; f < 8; ++f) {
      float r = fmaxf(S[0][f], S[1][f]);
#pragma unroll
      for (int off = 8; off >= 1; off >>= 1)
        r = fmaxf(r, __shfl_xor(r, off, 32));
      const float mnew = fmaxf(mrow[f], r);
      const float sc = __expf(mrow[f] - mnew);
      mrow[f] = mnew;
      const float p0 = (S[0][f] > -1e29f) ? __expf(S[0][f] - mnew) : 0.0f;
      const float p1 = (S[1][f] > -1e29f) ? __expf(S[1][f] - mnew) : 0.0f;
      S[0][f] = p0;
      S[1][f] = p1;
      float ps = p0 + p1;
#pragma unroll
      for (int off = 8; off >= 1; off >>= 1) ps += __shfl_xor(ps, off, 32);
      lrow[f] = lrow[f] * sc + ps;
      scale_f[f] = sc;
    }
#pragma unroll
    for (int d = 0; d < 16; ++d)
#pragma unroll
      for (int f = 0; f < 8; ++f) acc[d][f] *= scale_f[f];

    // ---- P: C-layout -> A-layout transpose through LDS ----
#pragma unroll
    for (int sub = 0; sub < 2; ++sub)
#pragma unroll
      for (int f = 0; f < 8; ++f)
        Pl[f + 8 * g][sub * 16 + ln] = (_Float16)S[sub][f];
    __syncthreads();
    const v16h pA = load_a_frag(&Pl[ln][0], g);
    __syncthreads();

    // ---- PV accumulation over all 16 dh tiles ----
#pragma unroll
    for (int d = 0; d < 16; ++d) acc[d] = wmma_f16(pA, vb[d], acc[d]);
  }

  // Normalize and store attn output as f16 [L][H*DH].
#pragma unroll
  for (int d = 0; d < 16; ++d)
#pragma unroll
    for (int f = 0; f < 8; ++f) {
      const int qrow = q0 + f + 8 * g;
      Oa[(size_t)qrow * (H_ * DH_) + h * DH_ + d * 16 + ln] =
          (_Float16)(acc[d][f] / lrow[f]);
    }
}

// ---------------------------------------------------------------------------
extern "C" void kernel_launch(void* const* d_in, const int* in_sizes, int n_in,
                              void* d_out, int out_size, void* d_ws,
                              size_t ws_size, hipStream_t stream) {
  (void)in_sizes; (void)n_in; (void)out_size; (void)ws_size;
  const float* X    = (const float*)d_in[0];
  const float* cosb = (const float*)d_in[1];
  const float* sinb = (const float*)d_in[2];
  // d_in[3] = attention_mask: recomputed analytically in-kernel.
  const float* Wq = (const float*)d_in[4];
  const float* Wk = (const float*)d_in[5];
  const float* Wv = (const float*)d_in[6];
  const float* Wo = (const float*)d_in[7];
  const float* qw = (const float*)d_in[8];
  const float* kw = (const float*)d_in[9];
  float* out = (float*)d_out;

  // f16 element counts
  const size_t nX  = (size_t)L_ * HS_;          // 10,485,760
  const size_t nWq = (size_t)H_ * DH_ * HS_;    // 10,485,760
  const size_t nWk = (size_t)HKV_ * DH_ * HS_;  //  5,242,880
  const size_t nWo = (size_t)HS_ * H_ * DH_;    // 10,485,760

  char* ws = (char*)d_ws;
  size_t off = 0;
  _Float16* Xh  = (_Float16*)(ws + off); off += nX * 2;
  _Float16* Wqh = (_Float16*)(ws + off); off += nWq * 2;
  _Float16* Wkh = (_Float16*)(ws + off); off += nWk * 2;
  _Float16* Wvh = (_Float16*)(ws + off); off += nWk * 2;
  _Float16* Woh = (_Float16*)(ws + off); off += nWo * 2;          // = 80 MiB
  _Float16* q16 = (_Float16*)(ws + off); off += (size_t)L_ * H_ * DH_ * 2;
  _Float16* k16 = (_Float16*)(ws + off); off += (size_t)L_ * HKV_ * DH_ * 2;
  _Float16* v16 = (_Float16*)(ws + off); off += (size_t)L_ * HKV_ * DH_ * 2;
  _Float16* Qf  = (_Float16*)(ws + off); off += (size_t)H_ * L_ * DH_ * 2;
  _Float16* Kf  = (_Float16*)(ws + off); off += (size_t)HKV_ * L_ * DH_ * 2;
  _Float16* Vt  = (_Float16*)(ws + off); off += (size_t)HKV_ * L_ * DH_ * 2;
  _Float16* Oa  = q16;  // reuse: q16 fully consumed by normrope before attn

  const dim3 wave(32);
  const dim3 b256(256);

  // 1) convert activations + weights to f16 (single streaming pass)
  cvt_f32_to_f16<<<dim3((unsigned)(nX  >> 10)), b256, 0, stream>>>(X,  Xh,  (int)nX);
  cvt_f32_to_f16<<<dim3((unsigned)(nWq >> 10)), b256, 0, stream>>>(Wq, Wqh, (int)nWq);
  cvt_f32_to_f16<<<dim3((unsigned)(nWk >> 10)), b256, 0, stream>>>(Wk, Wkh, (int)nWk);
  cvt_f32_to_f16<<<dim3((unsigned)(nWk >> 10)), b256, 0, stream>>>(Wv, Wvh, (int)nWk);
  cvt_f32_to_f16<<<dim3((unsigned)(nWo >> 10)), b256, 0, stream>>>(Wo, Woh, (int)nWo);

  // 2) QKV projections (f16 x f16 -> f16), 64x64 tile per wave
  gemm_f16<_Float16><<<dim3((H_ * DH_) / 64, L_ / 64), wave, 0, stream>>>(
      Xh, Wqh, q16, L_, H_ * DH_, HS_);
  gemm_f16<_Float16><<<dim3((HKV_ * DH_) / 64, L_ / 64), wave, 0, stream>>>(
      Xh, Wkh, k16, L_, HKV_ * DH_, HS_);
  gemm_f16<_Float16><<<dim3((HKV_ * DH_) / 64, L_ / 64), wave, 0, stream>>>(
      Xh, Wvh, v16, L_, HKV_ * DH_, HS_);

  // 3) RMS norm + RoPE, layout change for attention
  normrope_kernel<<<dim3(H_ + 2 * HKV_, L_), b256, 0, stream>>>(
      q16, k16, v16, cosb, sinb, qw, kw, Qf, Kf, Vt);

  // 4) sliding-window attention
  attn_kernel<<<dim3(L_ / 16, H_), wave, 0, stream>>>(Qf, Kf, Vt, Oa);

  // 5) output projection (f16 x f16 -> f32)
  gemm_f16<float><<<dim3(HS_ / 64, L_ / 64), wave, 0, stream>>>(
      Oa, Woh, out, L_, HS_, H_ * DH_);
}
